// Attention_94489281377
// MI455X (gfx1250) — matrix operand order repaired and checked by reading, not compiled, not run on hardware
//
#include <hip/hip_runtime.h>

// Causal attention B=16, S=2048, D=128, fp32 in/out.
// Round 2: prep kernel converts K -> f16 and V -> f16-transposed in d_ws;
// main kernel stages K/V tiles with the Tensor Data Mover (tensor_load_to_lds,
// padded 2D descriptors -> conflict-free LDS strides), double-buffered and
// overlapped with v_wmma_f32_16x16x32_f16 compute via TENSORcnt.

typedef __attribute__((ext_vector_type(16))) _Float16 v16h;
typedef __attribute__((ext_vector_type(8)))  _Float16 v8h;
typedef __attribute__((ext_vector_type(8)))  float    v8f;
typedef __attribute__((ext_vector_type(4)))  unsigned int u32x4;
typedef __attribute__((ext_vector_type(4)))  int          i32x4;
typedef __attribute__((ext_vector_type(8)))  int          i32x8;

namespace {
constexpr int   DK    = 128;
constexpr int   SEQ   = 2048;
constexpr int   BATCH = 16;
constexpr int   BM    = 64;    // queries per workgroup (16 per wave x 4 waves)
constexpr int   BN    = 32;    // keys per iteration
constexpr int   KLD   = 136;   // K tile LDS row stride (halves): 256B row + 16B pad
constexpr int   VLD   = 40;    // V^T tile LDS row stride (halves): 64B row + 16B pad
constexpr int   PLD   = 40;    // P staging row stride (halves)
constexpr float LOG2E = 1.4426950408889634f;
constexpr float SM_SCALE = 0.08838834764831845f * LOG2E;  // (1/sqrt(128))*log2e
constexpr float MASK_SUB = 1.0e10f;  // reference subtracts INF before scaling
}

// ---------------- prep: fp32 -> f16, V transposed to [b][d][s] ----------------
__launch_bounds__(128, 1)
__global__ void prep_kv(const float* __restrict__ Kg, const float* __restrict__ Vg,
                        _Float16* __restrict__ Kh, _Float16* __restrict__ Vth) {
  __shared__ __align__(16) _Float16 Vtile[32 * 136];
  const int tid = threadIdx.x;
  const int nst = SEQ / 32;
  const int b   = blockIdx.x / nst;
  const int s0  = (blockIdx.x - b * nst) * 32;
  const int r   = tid >> 2;         // 0..31 seq row in tile
  const int f0  = (tid & 3) * 32;   // feature chunk

  const float* krow = Kg + ((size_t)(b * SEQ + s0 + r)) * DK + f0;
  const float* vrow = Vg + ((size_t)(b * SEQ + s0 + r)) * DK + f0;
  _Float16*    kout = Kh + ((size_t)(b * SEQ + s0 + r)) * DK + f0;
#pragma unroll
  for (int j2 = 0; j2 < 4; ++j2) {
    v8h tk, tv;
#pragma unroll
    for (int e = 0; e < 8; ++e) {
      tk[e] = (_Float16)krow[8 * j2 + e];
      tv[e] = (_Float16)vrow[8 * j2 + e];
    }
    *(v8h*)(kout + 8 * j2) = tk;
    *(v8h*)&Vtile[r * 136 + f0 + 8 * j2] = tv;
  }
  __syncthreads();
  const int f = tid;  // 0..127 feature row
  _Float16* vout = Vth + (size_t)b * DK * SEQ + (size_t)f * SEQ + s0;
#pragma unroll
  for (int j2 = 0; j2 < 4; ++j2) {
    v8h t;
#pragma unroll
    for (int e = 0; e < 8; ++e) t[e] = Vtile[(8 * j2 + e) * 136 + f];
    *(v8h*)(vout + 8 * j2) = t;
  }
}

// ---------------- TDM 2D tile load: global f16 -> padded LDS ----------------
__device__ __forceinline__ void tdm_load_2d(unsigned lds_off, const _Float16* gptr,
                                            unsigned tensor_d0, unsigned tensor_d1,
                                            unsigned long long stride0,
                                            unsigned tile_d0, unsigned tile_d1,
                                            unsigned pad_interval, unsigned pad_amount) {
  const unsigned long long ga = (unsigned long long)(uintptr_t)gptr;
  u32x4 g0;
  g0[0] = 1u;                                   // count=1, user mode
  g0[1] = lds_off;                              // LDS byte address
  g0[2] = (unsigned)ga;                         // global_addr[31:0]
  g0[3] = (unsigned)((ga >> 32) & 0x01FFFFFFu)  // global_addr[56:32]
        | 0x80000000u;                          // type=2 ("image")
  i32x8 g1;
  g1[0] = (int)((1u << 16)                      // data_size = 2 bytes
              | (1u << 20)                      // pad_enable
              | (pad_interval << 22)
              | (pad_amount << 25));
  g1[1] = (int)((tensor_d0 & 0xFFFFu) << 16);                       // tensor_dim0[15:0]
  g1[2] = (int)((tensor_d0 >> 16) | ((tensor_d1 & 0xFFFFu) << 16)); // dim0 hi | dim1 lo
  g1[3] = (int)((tensor_d1 >> 16) | (tile_d0 << 16));               // dim1 hi | tile_dim0
  g1[4] = (int)(tile_d1 & 0xFFFFu);                                 // tile_dim1, tile_dim2=0
  g1[5] = (int)(unsigned)(stride0 & 0xFFFFFFFFull);                 // tensor_dim0_stride lo
  g1[6] = (int)(unsigned)((stride0 >> 32) & 0xFFFFull);             // stride hi, dim1_stride=0
  g1[7] = 0;
  const i32x4 z4 = {0, 0, 0, 0};
#if defined(__clang_major__) && (__clang_major__ >= 23)
  const i32x8 z8 = {0, 0, 0, 0, 0, 0, 0, 0};
  __builtin_amdgcn_tensor_load_to_lds(g0, g1, z4, z4, z8, 0);
#else
  __builtin_amdgcn_tensor_load_to_lds(g0, g1, z4, z4, 0);
#endif
}

__device__ __forceinline__ float half_rowmax(float v) {
#pragma unroll
  for (int off = 8; off >= 1; off >>= 1)
    v = fmaxf(v, __shfl_xor(v, off, 32));
  return v;
}
__device__ __forceinline__ float half_rowsum(float v) {
#pragma unroll
  for (int off = 8; off >= 1; off >>= 1)
    v += __shfl_xor(v, off, 32);
  return v;
}

// ---------------- main: FA2 with TDM double buffering ----------------
__launch_bounds__(128, 1)
__global__ void fa2_causal_tdm(const float* __restrict__ Qg,
                               const _Float16* __restrict__ Kh,
                               const _Float16* __restrict__ Vth,
                               float* __restrict__ Og) {
  __shared__ __align__(16) _Float16 Ks[2][BN * KLD];   // K tiles (key x feat), padded
  __shared__ __align__(16) _Float16 Vt[2][DK * VLD];   // V^T tiles (feat x key), padded
  __shared__ __align__(16) _Float16 Ps[4 * 16 * PLD];  // per-wave P staging

  const int tid  = threadIdx.x;
  const int wave = tid >> 5;
  const int lane = tid & 31;
  const int lh   = lane & 15;
  const int hi   = lane >> 4;

  const int nqt   = SEQ / BM;
  const int b     = blockIdx.x / nqt;
  const int qt    = blockIdx.x - b * nqt;
  const int qbase = qt * BM;
  const int q0    = qbase + wave * 16;

  // Q A-fragments straight into ISA layout, fp32 -> f16 once
  v16h qa[4];
  {
    const float* qrow = Qg + (size_t)(b * SEQ + q0 + lh) * DK;
#pragma unroll
    for (int c = 0; c < 4; ++c) {
      const float* p0 = qrow + 32 * c + 8 * hi;
      const float* p1 = qrow + 32 * c + 16 + 8 * hi;
      v16h a;
#pragma unroll
      for (int j = 0; j < 8; ++j) a[j]     = (_Float16)p0[j];
#pragma unroll
      for (int j = 0; j < 8; ++j) a[8 + j] = (_Float16)p1[j];
      qa[c] = a;
    }
  }

  v8f o_acc[8];
#pragma unroll
  for (int nt = 0; nt < 8; ++nt)
#pragma unroll
    for (int r = 0; r < 8; ++r) o_acc[nt][r] = 0.0f;

  float m_i[8], l_i[8];
#pragma unroll
  for (int r = 0; r < 8; ++r) { m_i[r] = -3.0e38f; l_i[r] = 0.0f; }

  const int nblocks = (qbase + BM) / BN;

  auto issue_tile = [&](int kb, int buf) {
    const int kkey = kb * BN;
    if (wave == 0) {
      // K tile: 32 rows x 128 halves, row 256B + 16B pad -> stride 136 halves
      tdm_load_2d((unsigned)(size_t)&Ks[buf][0],
                  Kh + (size_t)(b * SEQ + kkey) * DK,
                  DK, BN, (unsigned long long)DK, DK, BN,
                  /*pad_interval=*/5, /*pad_amount=*/3);
    } else if (wave == 1) {
      // V^T tile: 128 rows x 32 halves, row 64B + 16B pad -> stride 40 halves
      tdm_load_2d((unsigned)(size_t)&Vt[buf][0],
                  Vth + (size_t)b * DK * SEQ + kkey,
                  SEQ, DK, (unsigned long long)SEQ, BN, DK,
                  /*pad_interval=*/3, /*pad_amount=*/3);
    }
  };

  issue_tile(0, 0);

  for (int kb = 0; kb < nblocks; ++kb) {
    const int kkey = kb * BN;
    const int cur  = kb & 1;

    __syncthreads();  // everyone done reading buf[(kb+1)&1] from iteration kb-1
    if (kb + 1 < nblocks) {
      issue_tile(kb + 1, (kb + 1) & 1);              // overlap next DMA with compute
      __builtin_amdgcn_s_wait_tensorcnt(1);          // tile kb done, kb+1 in flight
    } else {
      __builtin_amdgcn_s_wait_tensorcnt(0);
    }
    __syncthreads();  // tile kb visible to all waves

    if (kkey <= q0 + 15) {  // wave-uniform causal skip
      // ---- S = Q K^T ----
      v8f sc[2];
#pragma unroll
      for (int t = 0; t < 2; ++t) {
        v8f s;
#pragma unroll
        for (int r = 0; r < 8; ++r) s[r] = 0.0f;
        const _Float16* kr = &Ks[cur][(t * 16 + lh) * KLD + 8 * hi];
#pragma unroll
        for (int c = 0; c < 4; ++c) {
          v16h bf;
          ((v8h*)&bf)[0] = *(const v8h*)(kr + 32 * c);
          ((v8h*)&bf)[1] = *(const v8h*)(kr + 32 * c + 16);
          s = __builtin_amdgcn_wmma_f32_16x16x32_f16(false, qa[c], false, bf,
                                                     (short)0, s, false, false);
        }
        // mask then scale (reference order), directly in exp2 domain
        const int kpos = kkey + t * 16 + lh;
#pragma unroll
        for (int r = 0; r < 8; ++r) {
          const int qpos = q0 + r + 8 * hi;
          float sv = s[r];
          sv = (kpos > qpos) ? (sv - MASK_SUB) : sv;
          s[r] = sv * SM_SCALE;
        }
        sc[t] = s;
      }

      // ---- online softmax (base-2) ----
#pragma unroll
      for (int r = 0; r < 8; ++r) {
        const float rm    = half_rowmax(fmaxf(sc[0][r], sc[1][r]));
        const float mnew  = fmaxf(m_i[r], rm);
        const float alpha = __builtin_amdgcn_exp2f(m_i[r] - mnew);
        const float p0    = __builtin_amdgcn_exp2f(sc[0][r] - mnew);
        const float p1    = __builtin_amdgcn_exp2f(sc[1][r] - mnew);
        const float rs    = half_rowsum(p0 + p1);
        l_i[r] = l_i[r] * alpha + rs;
        m_i[r] = mnew;
        const int M = r + 8 * hi;
        Ps[(wave * 16 + M) * PLD + lh]      = (_Float16)p0;
        Ps[(wave * 16 + M) * PLD + 16 + lh] = (_Float16)p1;
#pragma unroll
        for (int nt = 0; nt < 8; ++nt) o_acc[nt][r] *= alpha;
      }

      // ---- P back in A-fragment layout ----
      v16h pa;
      {
        const _Float16* pp = &Ps[(wave * 16 + lh) * PLD + 8 * hi];
        ((v8h*)&pa)[0] = *(const v8h*)pp;
        ((v8h*)&pa)[1] = *(const v8h*)(pp + 16);
      }

      // ---- O += P V ----
#pragma unroll
      for (int nt = 0; nt < 8; ++nt) {
        v16h vb;
        const _Float16* vp = &Vt[cur][(nt * 16 + lh) * VLD + 8 * hi];
        ((v8h*)&vb)[0] = *(const v8h*)vp;
        ((v8h*)&vb)[1] = *(const v8h*)(vp + 16);
        o_acc[nt] = __builtin_amdgcn_wmma_f32_16x16x32_f16(false, pa, false, vb,
                                                           (short)0, o_acc[nt],
                                                           false, false);
      }
    }
  }

  // ---- epilogue ----
#pragma unroll
  for (int nt = 0; nt < 8; ++nt) {
#pragma unroll
    for (int r = 0; r < 8; ++r) {
      const int qpos = q0 + r + 8 * hi;
      Og[(size_t)(b * SEQ + qpos) * DK + nt * 16 + lh] = o_acc[nt][r] / l_i[r];
    }
  }
}

extern "C" void kernel_launch(void* const* d_in, const int* in_sizes, int n_in,
                              void* d_out, int out_size, void* d_ws, size_t ws_size,
                              hipStream_t stream) {
  (void)in_sizes; (void)n_in; (void)out_size; (void)ws_size;
  const float* Q = (const float*)d_in[0];
  const float* K = (const float*)d_in[1];
  const float* V = (const float*)d_in[2];
  float*       O = (float*)d_out;

  _Float16* Kh  = (_Float16*)d_ws;                       // [B][S][D] f16
  _Float16* Vth = Kh + (size_t)BATCH * SEQ * DK;         // [B][D][S] f16

  prep_kv<<<dim3(BATCH * (SEQ / 32)), dim3(128), 0, stream>>>(K, V, Kh, Vth);
  fa2_causal_tdm<<<dim3(BATCH * (SEQ / BM)), dim3(128), 0, stream>>>(Q, Kh, Vth, O);
}
